// multihead_self_attention_13821204758577
// MI455X (gfx1250) — matrix operand order repaired and checked by reading, not compiled, not run on hardware
//
#include <hip/hip_runtime.h>

#define SEQ    2048
#define DMODEL 1024
#define HEADS  16
#define DKH    64
#define MTOT   4096   // B*S = 2*2048

typedef __attribute__((ext_vector_type(16))) __bf16 bf16x16;
typedef __attribute__((ext_vector_type(8)))  float  f32x8;

struct FragAB { uint4 lo, hi; };   // 8 VGPRs of packed bf16 (16 elems)

union FragCast { FragAB u; bf16x16 v; };
static_assert(sizeof(FragAB) == 32, "frag size");

static __device__ __forceinline__ bf16x16 frag_as_bf(const FragAB& f) {
  FragCast c; c.u = f; return c.v;
}

static __device__ __forceinline__ f32x8 wmma_bf16(const FragAB& a, const FragAB& b, f32x8 c) {
  return __builtin_amdgcn_wmma_f32_16x16x32_bf16(
      false, frag_as_bf(a), false, frag_as_bf(b), (short)0, c, false, false);
}

static __device__ __forceinline__ unsigned short f2bf(float x) {
  unsigned int u = __float_as_uint(x);
  u += 0x7FFFu + ((u >> 16) & 1u);     // round-to-nearest-even
  return (unsigned short)(u >> 16);
}

// A fragment (16x32, M x K): lane&15 = M row; lanes>=16 shift K by 8.
static __device__ __forceinline__ FragAB load_a(const unsigned short* A, int ld,
                                                int m0, int k0, int lane) {
  int m  = m0 + (lane & 15);
  int kk = k0 + ((lane & 16) ? 8 : 0);
  const unsigned short* p = A + (size_t)m * ld + kk;
  FragAB f;
  f.lo = *(const uint4*)(p);
  f.hi = *(const uint4*)(p + 16);
  return f;
}

// B fragment (32x16, K x N) from memory laid out W[n][k] (k contiguous):
// lane&15 = N col; lanes>=16 shift K by 8 (elems 0-7: K+0..7, elems 8-15: K+16..23).
static __device__ __forceinline__ FragAB load_b(const unsigned short* W, int ld,
                                                int n0, int k0, int lane) {
  int n  = n0 + (lane & 15);
  int kk = k0 + ((lane & 16) ? 8 : 0);
  const unsigned short* p = W + (size_t)n * ld + kk;
  FragAB f;
  f.lo = *(const uint4*)(p);
  f.hi = *(const uint4*)(p + 16);
  return f;
}

// 16-byte async copy global -> LDS (ASYNCcnt-tracked, CDNA5 async engine)
static __device__ __forceinline__ void async_ld16(const unsigned short* lds_ptr,
                                                  const unsigned short* g) {
  unsigned off = (unsigned)(size_t)lds_ptr;   // low 32 bits of flat LDS addr = LDS offset
  asm volatile("global_load_async_to_lds_b128 %0, %1, off"
               :: "v"(off), "v"(g) : "memory");
}

// ---------------- fp32 -> bf16 conversion ----------------
__global__ void f32_to_bf16_kernel(const float* __restrict__ src,
                                   unsigned short* __restrict__ dst, int n) {
  int i = blockIdx.x * blockDim.x + threadIdx.x;
  int stride = gridDim.x * blockDim.x;
  for (; i < n; i += stride) dst[i] = f2bf(src[i]);
}

// ---------------- NT GEMM: C[M,N] = A[M,K] * W[N,K]^T (bf16 in, f32 acc) -------------
// mode 0: N=3072 fused QKV epilogue -> Q [B,H,S,dk] (*1/8), K [B,H,S,dk], V^T [B,H,dk,S]
// mode 1: fp32 row-major write to Cout [M,N]
__global__ __launch_bounds__(256) void gemm_nt_kernel(
    const unsigned short* __restrict__ A, const unsigned short* __restrict__ W,
    int M, int N, int K, int mode,
    unsigned short* __restrict__ Qb, unsigned short* __restrict__ Kb,
    unsigned short* __restrict__ VTb, float* __restrict__ Cout) {
  int lane = threadIdx.x & 31;
  int wid  = threadIdx.x >> 5;          // 8 waves: 4 (M) x 2 (N)
  int m0 = blockIdx.y * 256 + (wid >> 1) * 64;
  int n0 = blockIdx.x * 128 + (wid & 1) * 64;
  int lh = (lane >> 4) & 1;             // lane half
  int ln = lane & 15;

  f32x8 acc[4][4] = {};
  for (int k0 = 0; k0 < K; k0 += 32) {
    FragAB af[4], bf[4];
#pragma unroll
    for (int i = 0; i < 4; ++i) af[i] = load_a(A, K, m0 + 16 * i, k0, lane);
#pragma unroll
    for (int j = 0; j < 4; ++j) bf[j] = load_b(W, K, n0 + 16 * j, k0, lane);
    if (k0 + 64 < K) {   // prefetch next-next k tile into caches (global_prefetch_b8)
#pragma unroll
      for (int i = 0; i < 4; ++i)
        __builtin_prefetch(A + (size_t)(m0 + 16 * i + ln) * K + k0 + 64 + lh * 8);
#pragma unroll
      for (int j = 0; j < 4; ++j)
        __builtin_prefetch(W + (size_t)(n0 + 16 * j + ln) * K + k0 + 64 + lh * 16);
    }
#pragma unroll
    for (int i = 0; i < 4; ++i)
#pragma unroll
      for (int j = 0; j < 4; ++j) acc[i][j] = wmma_bf16(af[i], bf[j], acc[i][j]);
  }

  if (mode == 1) {
#pragma unroll
    for (int i = 0; i < 4; ++i)
#pragma unroll
      for (int j = 0; j < 4; ++j) {
        f32x8 t = acc[i][j];
#pragma unroll
        for (int r = 0; r < 8; ++r)
          Cout[(size_t)(m0 + 16 * i + r + lh * 8) * N + (n0 + 16 * j + ln)] = t[r];
      }
  } else {
    int region = n0 >> 10;               // uniform per wave: 0=Q, 1=K, 2=V
    int nb = n0 & 1023;
#pragma unroll
    for (int i = 0; i < 4; ++i)
#pragma unroll
      for (int j = 0; j < 4; ++j) {
        f32x8 t = acc[i][j];
        int h = (nb + 16 * j) >> 6, d = ((nb + 16 * j) & 63) + ln;
#pragma unroll
        for (int r = 0; r < 8; ++r) {
          int m = m0 + 16 * i + r + lh * 8;
          int b = m >> 11, s = m & 2047;     // SEQ=2048
          size_t bh = (size_t)(b * 16 + h);
          if (region == 0)      Qb [(bh * SEQ + s) * DKH + d] = f2bf(t[r] * 0.125f);
          else if (region == 1) Kb [(bh * SEQ + s) * DKH + d] = f2bf(t[r]);
          else                  VTb[(bh * DKH + d) * SEQ + s] = f2bf(t[r]);
        }
      }
  }
}

// ---------------- causal flash attention --------------------------------------------
// Block = 8 waves = 128 query rows of one (b,h). K/V tiles (32 kv positions) are staged
// cooperatively into double-buffered LDS with async-to-LDS copies (ASYNCcnt), overlapped
// with WMMA compute on the previous tile.
__global__ __launch_bounds__(256) void flash_attn_kernel(
    const unsigned short* __restrict__ Qb, const unsigned short* __restrict__ Kb,
    const unsigned short* __restrict__ VTb, unsigned short* __restrict__ Ob) {
  __shared__ unsigned short sK[2][32 * 64];   // [kv][d]   4KB per buffer
  __shared__ unsigned short sV[2][64 * 32];   // [d][kv]   4KB per buffer
  __shared__ unsigned short sP[8][16 * 32];   // per-wave P tile

  int tid  = threadIdx.x;
  int lane = tid & 31;
  int wid  = tid >> 5;
  int bh = blockIdx.x >> 4;            // b*16+h
  int qb = blockIdx.x & 15;            // 128-row block within sequence
  int q0 = qb * 128 + wid * 16;        // this wave's query rows
  int lh = (lane >> 4) & 1;
  int ln = lane & 15;

  const unsigned short* Qh = Qb  + (size_t)bh * SEQ * DKH;
  const unsigned short* Kh = Kb  + (size_t)bh * SEQ * DKH;
  const unsigned short* Vh = VTb + (size_t)bh * DKH * SEQ;

  FragAB qf0 = load_a(Qh, DKH, q0, 0, lane);
  FragAB qf1 = load_a(Qh, DKH, q0, 32, lane);

  f32x8 o[4] = {};
  float mrow[8], lrow[8];
#pragma unroll
  for (int r = 0; r < 8; ++r) { mrow[r] = -1e30f; lrow[r] = 0.0f; }

  int kv_end = qb * 128 + 128;         // block-uniform causal bound

  // cooperative stage of one 32-wide kv tile: 256 threads x (16B K + 16B V)
  auto stage = [&](int buf, int kv0) {
    int krow = tid >> 3, kc = (tid & 7) * 8;   // K: 32 x 64
    async_ld16(&sK[buf][tid * 8], Kh + (size_t)(kv0 + krow) * DKH + kc);
    int vrow = tid >> 2, vc = (tid & 3) * 8;   // V^T: 64 x 32
    async_ld16(&sV[buf][tid * 8], Vh + (size_t)vrow * SEQ + kv0 + vc);
  };

  stage(0, 0);                                   // prologue: tile 0 in flight

  for (int kv0 = 0, blk = 0; kv0 < kv_end; kv0 += 32, ++blk) {
    int cur = blk & 1;
    __syncthreads();                             // readers of buf cur^1 are done
    if (kv0 + 32 < kv_end) {
      stage(cur ^ 1, kv0 + 32);                  // next tile in flight
      asm volatile("s_wait_asynccnt 2" ::: "memory");   // current tile landed
    } else {
      asm volatile("s_wait_asynccnt 0" ::: "memory");
    }
    __syncthreads();                             // all waves' copies visible

    if (kv0 < q0 + 16) {                         // wave-uniform causal skip
      const unsigned short* tK = &sK[cur][0];
      const unsigned short* tV = &sV[cur][0];
      // scores: S[16 x 32] = Q(16x64) * K^T(64x32), K from LDS (ds_load_b128)
      f32x8 s0 = {}, s1 = {};
      s0 = wmma_bf16(qf0, load_b(tK, DKH, 0,  0,  lane), s0);
      s0 = wmma_bf16(qf1, load_b(tK, DKH, 0,  32, lane), s0);
      s1 = wmma_bf16(qf0, load_b(tK, DKH, 16, 0,  lane), s1);
      s1 = wmma_bf16(qf1, load_b(tK, DKH, 16, 32, lane), s1);

      float corr[8], mnew[8];
#pragma unroll
      for (int r = 0; r < 8; ++r) {
        int row = q0 + r + lh * 8;
        if (kv0 + ln      > row) s0[r] = -1e30f;
        if (kv0 + 16 + ln > row) s1[r] = -1e30f;
        float v = fmaxf(s0[r], s1[r]);
        v = fmaxf(v, __shfl_xor(v, 1));
        v = fmaxf(v, __shfl_xor(v, 2));
        v = fmaxf(v, __shfl_xor(v, 4));
        v = fmaxf(v, __shfl_xor(v, 8));
        mnew[r] = fmaxf(mrow[r], v);
        corr[r] = __expf(mrow[r] - mnew[r]);
        mrow[r] = mnew[r];
        float p0 = __expf(s0[r] - mnew[r]);
        float p1 = __expf(s1[r] - mnew[r]);
        float rs = p0 + p1;
        rs += __shfl_xor(rs, 1);
        rs += __shfl_xor(rs, 2);
        rs += __shfl_xor(rs, 4);
        rs += __shfl_xor(rs, 8);
        lrow[r] = lrow[r] * corr[r] + rs;
        sP[wid][(r + lh * 8) * 32 + ln]      = f2bf(p0);   // D-layout -> [row][col]
        sP[wid][(r + lh * 8) * 32 + 16 + ln] = f2bf(p1);
      }
#pragma unroll
      for (int t = 0; t < 4; ++t)
#pragma unroll
        for (int r = 0; r < 8; ++r) o[t][r] *= corr[r];

      // wave-private LDS round-trip; DS is in-order per wave
      asm volatile("s_wait_dscnt 0" ::: "memory");

      const unsigned short* pp = &sP[wid][0];
      FragAB pf;
      {
        int kk = lh ? 8 : 0;
        pf.lo = *(const uint4*)(pp + ln * 32 + kk);
        pf.hi = *(const uint4*)(pp + ln * 32 + 16 + kk);
      }
      // O += P(16x32) * V(32x64), V from LDS
#pragma unroll
      for (int t = 0; t < 4; ++t)
        o[t] = wmma_bf16(pf, load_b(tV, 32, t * 16, 0, lane), o[t]);
    }
  }

  int b = bh >> 4, h = bh & 15;
#pragma unroll
  for (int t = 0; t < 4; ++t)
#pragma unroll
    for (int r = 0; r < 8; ++r) {
      int row = q0 + r + lh * 8;
      float v = o[t][r] / lrow[r];
      Ob[((size_t)(b * SEQ + row)) * DMODEL + h * 64 + t * 16 + ln] = f2bf(v);
    }
}

extern "C" void kernel_launch(void* const* d_in, const int* in_sizes, int n_in,
                              void* d_out, int out_size, void* d_ws, size_t ws_size,
                              hipStream_t stream) {
  const float* x  = (const float*)d_in[0];
  const float* Wq = (const float*)d_in[1];
  const float* Wk = (const float*)d_in[2];
  const float* Wv = (const float*)d_in[3];
  const float* Wo = (const float*)d_in[4];

  unsigned short* Xbf  = (unsigned short*)d_ws;
  unsigned short* Wqkv = Xbf  + (size_t)MTOT * DMODEL;          // 3 stacked [1024,1024]
  unsigned short* Wob  = Wqkv + (size_t)3 * DMODEL * DMODEL;
  unsigned short* Qb   = Wob  + (size_t)DMODEL * DMODEL;        // [B,H,S,dk]
  unsigned short* Kb   = Qb   + (size_t)MTOT * DMODEL;
  unsigned short* VTb  = Kb   + (size_t)MTOT * DMODEL;          // [B,H,dk,S]
  unsigned short* Ob   = VTb  + (size_t)MTOT * DMODEL;          // [B*S, H*dk]

  int nx = MTOT * DMODEL, nw = DMODEL * DMODEL;
  f32_to_bf16_kernel<<<1024, 256, 0, stream>>>(x,  Xbf, nx);
  f32_to_bf16_kernel<<<512, 256, 0, stream>>>(Wq, Wqkv,          nw);
  f32_to_bf16_kernel<<<512, 256, 0, stream>>>(Wk, Wqkv + nw,     nw);
  f32_to_bf16_kernel<<<512, 256, 0, stream>>>(Wv, Wqkv + 2 * nw, nw);
  f32_to_bf16_kernel<<<512, 256, 0, stream>>>(Wo, Wob, nw);

  gemm_nt_kernel<<<dim3(3072 / 128, MTOT / 256), 256, 0, stream>>>(
      Xbf, Wqkv, MTOT, 3072, DMODEL, 0, Qb, Kb, VTb, nullptr);

  flash_attn_kernel<<<512, 256, 0, stream>>>(Qb, Kb, VTb, Ob);

  gemm_nt_kernel<<<dim3(DMODEL / 128, MTOT / 256), 256, 0, stream>>>(
      Ob, Wob, MTOT, DMODEL, DMODEL, 1, nullptr, nullptr, nullptr, (float*)d_out);
}
